// FlashMultiheadSelfAttention_2113123910292
// MI455X (gfx1250) — compile-verified, matrix operand order checked by reading, not run on hardware
//
#include <hip/hip_runtime.h>
#include <hip/hip_bf16.h>

typedef __attribute__((ext_vector_type(8)))  float        v8f;
typedef __attribute__((ext_vector_type(8)))  __bf16       v8bf;
typedef __attribute__((ext_vector_type(16))) __bf16       v16bf;
typedef __attribute__((ext_vector_type(4)))  unsigned int u32x4;
typedef __attribute__((ext_vector_type(8)))  int          i32x8;
typedef __attribute__((ext_vector_type(4)))  int          i32x4;

#define TILE 128
#define KT   32
#define LDSLD 40   // padded row pitch (bf16 elems): conflict-free, 16B-aligned

#if defined(__HIP_DEVICE_COMPILE__) && __has_builtin(__builtin_amdgcn_tensor_load_to_lds) && __has_builtin(__builtin_amdgcn_s_wait_tensorcnt)
#define TDM_OK 1
#else
#define TDM_OK 0
#endif

template <class T, class U> struct same_t { static constexpr bool v = false; };
template <class T> struct same_t<T, T>    { static constexpr bool v = true;  };

// ---------- TDM: DMA one 128-row x 32-col bf16 tile into LDS ----------------
// D# per ISA 8.3-8.5: count=1, type=2(image), data_size=2B, gather off,
// pad_enable: every 16 DWORDs (=32 bf16 row) insert 4 DWORDs -> LDS pitch 40.
// This toolchain's builtin is the 6-arg clang-23 form:
//   (uint32x4 g0, int32x8 g1, int32x4 g2, int32x4 g3, int32x8 gx, int cpol)
__device__ __forceinline__ void tdm_load_2d(const __bf16* tile, unsigned lds_off,
                                            int ld_elems) {
#if TDM_OK
    unsigned long long ga = (unsigned long long)(uintptr_t)tile;
    u32x4 g0;
    g0[0] = 1u;                                            // count=1, user-mode
    g0[1] = lds_off;                                       // lds_addr (bytes)
    g0[2] = (unsigned)ga;                                  // global_addr[31:0]
    g0[3] = (unsigned)((ga >> 32) & 0x1FFFFFFu) | (2u << 30);  // [56:32] | type=2
    i32x8 g1;
    g1[0] = (int)((1u << 16)        // data_size = 2 bytes
                | (1u << 20)        // pad_enable
                | (3u << 22)        // pad_interval: 16 DWORDs
                | (3u << 25));      // pad_amount:   4 DWORDs
    g1[1] = (int)(((unsigned)ld_elems & 0xFFFFu) << 16);   // tensor_dim0 lo16
    g1[2] = (int)((((unsigned)ld_elems >> 16) & 0xFFFFu)   // tensor_dim0 hi16
                | (128u << 16));                           // tensor_dim1 = 128
    g1[3] = (int)((unsigned)KT << 16);                     // tile_dim0 = 32
    g1[4] = 128;                                           // tile_dim1=128, tile_dim2=0
    g1[5] = ld_elems;                                      // tensor_dim0_stride lo32
    g1[6] = 0;
    g1[7] = 0;
    i32x4 gz4 = (i32x4)0;                                  // groups 2/3 unused (2D)
    i32x8 gz8 = (i32x8)0;
    __builtin_amdgcn_tensor_load_to_lds(g0, g1, gz4, gz4, gz8, 0);
#else
    (void)tile; (void)lds_off; (void)ld_elems;
#endif
}
__device__ __forceinline__ void wait_tensor0() {
#if TDM_OK
    __builtin_amdgcn_s_wait_tensorcnt(0);
#endif
}

// ---------- manual tile staging: 16 elems per thread-row-chunk --------------
__device__ __forceinline__ void load_row16(const float* __restrict__ src, __bf16* dst) {
    const float4* p = (const float4*)src;
    float4 f0 = p[0], f1 = p[1], f2 = p[2], f3 = p[3];
    v8bf v0, v1;
    v0[0]=(__bf16)f0.x; v0[1]=(__bf16)f0.y; v0[2]=(__bf16)f0.z; v0[3]=(__bf16)f0.w;
    v0[4]=(__bf16)f1.x; v0[5]=(__bf16)f1.y; v0[6]=(__bf16)f1.z; v0[7]=(__bf16)f1.w;
    v1[0]=(__bf16)f2.x; v1[1]=(__bf16)f2.y; v1[2]=(__bf16)f2.z; v1[3]=(__bf16)f2.w;
    v1[4]=(__bf16)f3.x; v1[5]=(__bf16)f3.y; v1[6]=(__bf16)f3.z; v1[7]=(__bf16)f3.w;
    *(v8bf*)dst       = v0;
    *(v8bf*)(dst + 8) = v1;
}
__device__ __forceinline__ void load_row16(const __bf16* __restrict__ src, __bf16* dst) {
    v8bf a = ((const v8bf*)src)[0];
    v8bf b = ((const v8bf*)src)[1];
    *(v8bf*)dst       = a;
    *(v8bf*)(dst + 8) = b;
}

__device__ __forceinline__ void cstore(float*  p, float v) { *p = v; }
__device__ __forceinline__ void cstore(__bf16* p, float v) { *p = (__bf16)v; }

// one K-step of WMMA work for this wave from staged LDS tiles
__device__ __forceinline__ void mma_stage(const __bf16* As, const __bf16* Bs,
                                          v8f acc[2][4], int wm, int wn,
                                          int l16, int half) {
    v16bf afr[2], bfr[4];
#pragma unroll
    for (int mt = 0; mt < 2; ++mt) {
        int r = wm * 32 + mt * 16 + l16;
        v8bf lo = *(const v8bf*)&As[r * LDSLD + half * 8];
        v8bf hi = *(const v8bf*)&As[r * LDSLD + 16 + half * 8];
        afr[mt] = __builtin_shufflevector(lo, hi, 0,1,2,3,4,5,6,7,8,9,10,11,12,13,14,15);
    }
#pragma unroll
    for (int nt = 0; nt < 4; ++nt) {
        int r = wn * 64 + nt * 16 + l16;
        v8bf lo = *(const v8bf*)&Bs[r * LDSLD + half * 8];
        v8bf hi = *(const v8bf*)&Bs[r * LDSLD + 16 + half * 8];
        bfr[nt] = __builtin_shufflevector(lo, hi, 0,1,2,3,4,5,6,7,8,9,10,11,12,13,14,15);
    }
#pragma unroll
    for (int mt = 0; mt < 2; ++mt)
#pragma unroll
        for (int nt = 0; nt < 4; ++nt)
            acc[mt][nt] = __builtin_amdgcn_wmma_f32_16x16x32_bf16(
                false, afr[mt], false, bfr[nt], (short)0, acc[mt][nt], false, false);
}

// ---------- generic C = A * B^T GEMM on v_wmma_f32_16x16x32_bf16 ------------
// A: M x K row-major; B: N x K row-major; C: M x N row-major.
// MODE 0: dense. MODE 1: causal score GEMM (skip fully-masked tiles).
// MODE 2: PV GEMM (A cols > row are zero -> clip k-loop at diagonal).
// bf16 x bf16 instantiations use TDM double-buffered DMA staging.
template <typename AT, typename BT, typename OT, int MODE>
__global__ __launch_bounds__(256) void gemm_tn(
    const AT* __restrict__ A, const BT* __restrict__ B, OT* __restrict__ C,
    int M, int N, int K, float scale)
{
    __shared__ __align__(16) __bf16 As[2][TILE * LDSLD];
    __shared__ __align__(16) __bf16 Bs[2][TILE * LDSLD];

    const int m0 = blockIdx.y * TILE;
    const int n0 = blockIdx.x * TILE;
    if (MODE == 1 && n0 > m0 + (TILE - 1)) return;   // fully masked tile

    const size_t bz = blockIdx.z;
    A += bz * (size_t)M * K;
    B += bz * (size_t)N * K;
    C += bz * (size_t)M * N;

    const int tid  = threadIdx.x;
    const int lane = tid & 31;
    const int wave = tid >> 5;          // 8 waves
    const int wm   = wave & 3;          // 4 row groups of 32
    const int wn   = wave >> 2;         // 2 col groups of 64
    const int l16  = lane & 15;
    const int half = lane >> 4;         // K 0-7,16-23 | 8-15,24-31

    v8f acc[2][4];
#pragma unroll
    for (int i = 0; i < 2; ++i)
#pragma unroll
        for (int j = 0; j < 4; ++j) acc[i][j] = (v8f)0.0f;

    int kEnd = K;
    if (MODE == 2) { int lim = m0 + TILE; kEnd = lim < K ? lim : K; }

    constexpr bool TDMP = TDM_OK && same_t<AT, __bf16>::v && same_t<BT, __bf16>::v;

    if constexpr (TDMP) {
        // ---- TDM double-buffered pipeline: DMA stage s+1 while computing s
        const unsigned ldsA = (unsigned)(uintptr_t)&As[0][0];
        const unsigned ldsB = (unsigned)(uintptr_t)&Bs[0][0];
        const unsigned bufB = TILE * LDSLD * 2u;   // bytes per buffer
        const int nStages = kEnd / KT;
        if (wave == 0) {
            tdm_load_2d((const __bf16*)A + (size_t)m0 * K, ldsA, K);
            tdm_load_2d((const __bf16*)B + (size_t)n0 * K, ldsB, K);
        }
        for (int s = 0; s < nStages; ++s) {
            if (wave == 0) wait_tensor0();   // stage s resident in LDS
            __syncthreads();                 // all see it; buf (s+1)&1 is free
            if (wave == 0 && s + 1 < nStages) {
                int kn = (s + 1) * KT, bn = (s + 1) & 1;
                tdm_load_2d((const __bf16*)A + (size_t)m0 * K + kn, ldsA + bn * bufB, K);
                tdm_load_2d((const __bf16*)B + (size_t)n0 * K + kn, ldsB + bn * bufB, K);
            }
            mma_stage(&As[s & 1][0], &Bs[s & 1][0], acc, wm, wn, l16, half);
        }
    } else {
        // ---- manual staging (fp32 sources need VALU convert anyway)
        const int grow = tid >> 1;          // tile row staged by this thread
        const int gcg  = (tid & 1) * 16;    // k-chunk 0 | 16
        for (int k0 = 0; k0 < kEnd; k0 += KT) {
            const AT* ga = A + (size_t)(m0 + grow) * K + k0 + gcg;
            const BT* gb = B + (size_t)(n0 + grow) * K + k0 + gcg;
            __syncthreads();
            load_row16(ga, &As[0][grow * LDSLD + gcg]);
            load_row16(gb, &Bs[0][grow * LDSLD + gcg]);
            __syncthreads();
            mma_stage(&As[0][0], &Bs[0][0], acc, wm, wn, l16, half);
        }
    }

#pragma unroll
    for (int mt = 0; mt < 2; ++mt)
#pragma unroll
        for (int nt = 0; nt < 4; ++nt) {
            int col   = n0 + wn * 64 + nt * 16 + l16;
            int rbase = m0 + wm * 32 + mt * 16 + half * 8;
#pragma unroll
            for (int r = 0; r < 8; ++r)
                cstore(&C[(size_t)(rbase + r) * N + col], acc[mt][nt][r] * scale);
        }
}

// ---------- RoPE (interleaved pairs) + fp32 -> bf16 -------------------------
__global__ __launch_bounds__(256) void rope_to_bf16(
    const float* __restrict__ in, __bf16* __restrict__ out,
    const int* __restrict__ pos, int seq, int total)
{
    int idx = blockIdx.x * 256 + threadIdx.x;
    if (idx >= total) return;                 // total = B*seq*1024 pairs
    int bn = idx >> 10;
    int pp = idx & 1023;
    int q  = pp & 63;
    int o1 = (pp >> 6) * 128 + q * 2;
    int ni = bn & (seq - 1);
    float p    = (float)pos[ni];
    float freq = __expf(-(2.0f * q / 128.0f) * 9.210340371976184f); // theta=1e4
    float sn, cs;
    __sincosf(p * freq, &sn, &cs);
    const float* r = in + (size_t)bn * 2048 + o1;
    float x1 = r[0], x2 = r[1];
    __bf16* w = out + (size_t)bn * 2048 + o1;
    w[0] = (__bf16)(cs * x1 - sn * x2);
    w[1] = (__bf16)(sn * x1 + cs * x2);
}

// ---------- bf16 square transpose (per batch) -------------------------------
__global__ __launch_bounds__(256) void transpose_bf16(
    const __bf16* __restrict__ in, __bf16* __restrict__ out, int n)
{
    __shared__ __bf16 tile[32][33];
    size_t base = (size_t)blockIdx.z * n * n;
    int bx = blockIdx.x * 32, by = blockIdx.y * 32;
    int tx = threadIdx.x, ty = threadIdx.y;   // 32 x 8
#pragma unroll
    for (int i = 0; i < 32; i += 8)
        tile[ty + i][tx] = in[base + (size_t)(by + ty + i) * n + bx + tx];
    __syncthreads();
#pragma unroll
    for (int i = 0; i < 32; i += 8)
        out[base + (size_t)(bx + ty + i) * n + by + tx] = tile[tx][ty + i];
}

// ---------- causal row softmax: fp32 scores -> bf16 probabilities -----------
__global__ __launch_bounds__(256) void softmax_causal(
    const float* __restrict__ S, __bf16* __restrict__ P, int n)
{
    int row = blockIdx.x;                 // 0 .. B*n-1
    int i   = row & (n - 1);              // query position
    const float* s = S + (size_t)row * n;
    __bf16*      p = P + (size_t)row * n;
    __shared__ float red[256];
    int tid = threadIdx.x;

    float m = -3.4e38f;
    for (int j = tid; j <= i; j += 256) m = fmaxf(m, s[j]);
    red[tid] = m; __syncthreads();
    for (int w = 128; w > 0; w >>= 1) { if (tid < w) red[tid] = fmaxf(red[tid], red[tid + w]); __syncthreads(); }
    m = red[0]; __syncthreads();

    float sum = 0.0f;
    for (int j = tid; j <= i; j += 256) sum += __expf(s[j] - m);
    red[tid] = sum; __syncthreads();
    for (int w = 128; w > 0; w >>= 1) { if (tid < w) red[tid] += red[tid + w]; __syncthreads(); }
    float inv = 1.0f / red[0];

    for (int j = tid; j < n; j += 256)
        p[j] = (__bf16)((j <= i) ? __expf(s[j] - m) * inv : 0.0f);
}

// ---------------------------------------------------------------------------
extern "C" void kernel_launch(void* const* d_in, const int* in_sizes, int n_in,
                              void* d_out, int out_size, void* d_ws, size_t ws_size,
                              hipStream_t stream) {
    (void)in_sizes; (void)n_in; (void)out_size; (void)ws_size;
    const float* x   = (const float*)d_in[0];
    const int*   pos = (const int*)d_in[1];
    const float* wq  = (const float*)d_in[2];
    const float* wk  = (const float*)d_in[3];
    const float* wv  = (const float*)d_in[4];
    const float* wo  = (const float*)d_in[5];
    float* out = (float*)d_out;

    const int BT = 4, S = 2048, D = 2048;
    const size_t matE = (size_t)BT * S * D;

    char* ws = (char*)d_ws;
    float*  R0  = (float*) (ws);                  // fp32 scratch (Qraw/Kraw, S)
    float*  R1  = (float*) (ws + matE * 4);       // fp32 scratch (O)
    __bf16* Rq  = (__bf16*)(ws + matE * 8);
    __bf16* Rk  = (__bf16*)(ws + matE * 10);
    __bf16* Rv  = (__bf16*)(ws + matE * 12);
    __bf16* Rvt = (__bf16*)(ws + matE * 14);
    __bf16* Rp  = (__bf16*)(ws + matE * 16);      // total 18*matE ~ 302 MB

    dim3 blk(256);
    dim3 gProj(D / TILE, (BT * S) / TILE, 1);     // (16, 64)
    dim3 gAttn(S / TILE, S / TILE, BT);           // (16, 16, 4)
    const float iscale = 0.022097086912079608f;   // 1/sqrt(2048)
    const int nPairs = BT * S * 1024;

    // Q = x Wq^T ; RoPE -> bf16
    gemm_tn<float, float, float, 0><<<gProj, blk, 0, stream>>>(x, wq, R0, BT * S, D, D, 1.0f);
    rope_to_bf16<<<(nPairs + 255) / 256, 256, 0, stream>>>(R0, Rq, pos, S, nPairs);
    // K = x Wk^T ; RoPE -> bf16
    gemm_tn<float, float, float, 0><<<gProj, blk, 0, stream>>>(x, wk, R0, BT * S, D, D, 1.0f);
    rope_to_bf16<<<(nPairs + 255) / 256, 256, 0, stream>>>(R0, Rk, pos, S, nPairs);
    // V = x Wv^T (bf16) ; transpose for PV GEMM
    gemm_tn<float, float, __bf16, 0><<<gProj, blk, 0, stream>>>(x, wv, Rv, BT * S, D, D, 1.0f);
    transpose_bf16<<<dim3(S / 32, S / 32, BT), dim3(32, 8), 0, stream>>>(Rv, Rvt, S);
    // S = scale * Q K^T (causal tile skip), batched — TDM pipeline
    gemm_tn<__bf16, __bf16, float, 1><<<gAttn, blk, 0, stream>>>(Rq, Rk, R0, S, S, D, iscale);
    // P = causal softmax(S) -> bf16
    softmax_causal<<<BT * S, 256, 0, stream>>>(R0, Rp, S);
    // O = P V (= P * (V^T)^T, causal k clip), batched — TDM pipeline
    gemm_tn<__bf16, __bf16, float, 2><<<gAttn, blk, 0, stream>>>(Rp, Rvt, R1, S, S, S, 1.0f);
    // out = O Wo^T
    gemm_tn<float, float, float, 0><<<gProj, blk, 0, stream>>>(R1, wo, out, BT * S, D, D, 1.0f);
}
